// MultiHeadAttention_GQA_48335561949481
// MI455X (gfx1250) — matrix operand order repaired and checked by reading, not compiled, not run on hardware
//
#include <hip/hip_runtime.h>

// ---------------------------------------------------------------------------
// GQA attention for gfx1250 (MI455X), all matmuls via v_wmma_f32_16x16x32_bf16.
// B=2, S=2048, DM=2048, H=32, G=8, D=64, GS=4.
// ---------------------------------------------------------------------------

typedef __bf16 bf16;
typedef __attribute__((ext_vector_type(8)))  bf16  bf16x8;
typedef __attribute__((ext_vector_type(16))) bf16  v16bf;
typedef __attribute__((ext_vector_type(8)))  float v8f;

#define DEV __device__ __forceinline__

enum : int { kB = 2, kS = 2048, kDM = 2048, kH = 32, kG = 8, kD = 64,
             kM = kB * kS, kHD = kH * kD, kGD = kG * kD };

DEV v16bf cat8(bf16x8 lo, bf16x8 hi) {
  return __builtin_shufflevector(lo, hi, 0,1,2,3,4,5,6,7,8,9,10,11,12,13,14,15);
}
// B operand: 16 contiguous bf16 (lane = output column, K chunk = 16*lanehalf)
DEV v16bf load_fragB(const bf16* p) {
  bf16x8 lo = *(const bf16x8*)p;
  bf16x8 hi = *(const bf16x8*)(p + 8);
  return cat8(lo, hi);
}
// A operand: lane = row; 16 contiguous K at k0+16*lh (matches B operand order)
DEV v16bf load_fragA(const bf16* row, int k0, int lh) {
  bf16x8 lo = *(const bf16x8*)(row + k0 + 16 * lh);
  bf16x8 hi = *(const bf16x8*)(row + k0 + 16 * lh + 8);
  return cat8(lo, hi);
}
DEV v8f wmma_bf16(v16bf a, v16bf b, v8f c) {
  return __builtin_amdgcn_wmma_f32_16x16x32_bf16(false, a, false, b, (short)0, c,
                                                 false, false);
}

// ---------------------------------------------------------------------------
// fp32 -> bf16 bulk convert (x and attention activations stay bf16 in L2)
// ---------------------------------------------------------------------------
__global__ __launch_bounds__(256) void k_cvt(const float* __restrict__ src,
                                             bf16* __restrict__ dst, int n) {
  int i = (blockIdx.x * 256 + threadIdx.x) * 4;
  if (i + 3 < n) {
    float4 v = *(const float4*)(src + i);
    dst[i + 0] = (bf16)v.x;
    dst[i + 1] = (bf16)v.y;
    dst[i + 2] = (bf16)v.z;
    dst[i + 3] = (bf16)v.w;
  }
}

// ---------------------------------------------------------------------------
// LDS-tiled transpose: src fp32 [K][N] -> dst bf16 [N][K]
// (weights go to [N][K] so WMMA B-operand loads are contiguous b128s)
// ---------------------------------------------------------------------------
__global__ __launch_bounds__(256) void k_transpose(const float* __restrict__ src,
                                                   bf16* __restrict__ dst,
                                                   int K, int N) {
  __shared__ float tile[32][33];
  int kb = blockIdx.y * 32, nb = blockIdx.x * 32;
  int tx = threadIdx.x, ty = threadIdx.y;
#pragma unroll
  for (int j = 0; j < 4; ++j)
    tile[ty + j * 8][tx] = src[(size_t)(kb + ty + j * 8) * N + nb + tx];
  __syncthreads();
#pragma unroll
  for (int j = 0; j < 4; ++j)
    dst[(size_t)(nb + ty + j * 8) * K + kb + tx] = (bf16)tile[tx][ty + j * 8];
}

// ---------------------------------------------------------------------------
// GEMM: out[M,N] = A[M,K](bf16,row-major) * Bt[N,K](bf16) + bias
// wave tile = 32Mx64N (2 A frags reuse each B frag), block = 4 waves
// (128Mx64N per block), K step 32: 12 b128 loads per 8 WMMAs.
// MODE 0: bf16 row-major out; MODE 1: bf16 transposed out [N][M] (for V^T);
// MODE 2: fp32 row-major out (final projection).
// ---------------------------------------------------------------------------
template <int MODE>
__global__ __launch_bounds__(128) void k_gemm(const bf16* __restrict__ A,
                                              const bf16* __restrict__ Bt,
                                              const float* __restrict__ bias,
                                              void* __restrict__ outp,
                                              int M, int N, int K) {
  int lane = threadIdx.x & 31;
  int wave = threadIdx.x >> 5;
  int lh = lane >> 4, ln = lane & 15;
  int mbase = (blockIdx.y * 4 + wave) * 32;
  int nbase = blockIdx.x * 64;

  v8f acc[2][4];
#pragma unroll
  for (int ms = 0; ms < 2; ++ms)
#pragma unroll
    for (int t = 0; t < 4; ++t)
#pragma unroll
      for (int r = 0; r < 8; ++r) acc[ms][t][r] = 0.0f;

  const bf16* arow0 = A + (size_t)(mbase + ln) * K;
  const bf16* arow1 = arow0 + (size_t)16 * K;
  const bf16* brow0 = Bt + (size_t)(nbase + ln) * K + 16 * lh;

  for (int k0 = 0; k0 < K; k0 += 32) {
    if (k0 + 64 < K) {  // gfx1250 global_prefetch of the next-next A chunk
      __builtin_prefetch(arow0 + k0 + 64, 0, 0);
      __builtin_prefetch(arow1 + k0 + 64, 0, 0);
    }
    v16bf af0 = load_fragA(arow0, k0, lh);
    v16bf af1 = load_fragA(arow1, k0, lh);
#pragma unroll
    for (int nt = 0; nt < 4; ++nt) {
      v16bf bfrag = load_fragB(brow0 + (size_t)nt * 16 * K + k0);
      acc[0][nt] = wmma_bf16(af0, bfrag, acc[0][nt]);
      acc[1][nt] = wmma_bf16(af1, bfrag, acc[1][nt]);
    }
  }

#pragma unroll
  for (int ms = 0; ms < 2; ++ms) {
    int mb = mbase + ms * 16;
#pragma unroll
    for (int nt = 0; nt < 4; ++nt) {
      int n = nbase + nt * 16 + ln;
      float bv = bias[n];
#pragma unroll
      for (int r = 0; r < 8; ++r) acc[ms][nt][r] += bv;

      if (MODE == 0) {                      // bf16 [M][N]
        bf16* o = (bf16*)outp;
#pragma unroll
        for (int r = 0; r < 8; ++r)
          o[(size_t)(mb + 8 * lh + r) * N + n] = (bf16)acc[ms][nt][r];
      } else if (MODE == 1) {               // bf16 [N][M]  (transposed: V^T)
        bf16x8 pk;
#pragma unroll
        for (int r = 0; r < 8; ++r) pk[r] = (bf16)acc[ms][nt][r];
        *(bf16x8*)((bf16*)outp + (size_t)n * M + mb + 8 * lh) = pk;
      } else {                              // fp32 [M][N]
        float* o = (float*)outp;
#pragma unroll
        for (int r = 0; r < 8; ++r)
          o[(size_t)(mb + 8 * lh + r) * N + n] = acc[ms][nt][r];
      }
    }
  }
}

// ---------------------------------------------------------------------------
// Flash-style causal GQA attention.
// One wave owns a 16-query tile of TWO heads sharing one KV group (GQA reuse):
// every K/V A-fragment feeds two WMMAs -> ~1 b128 load per WMMA, half the
// K/V L2 traffic. Computes S^T = K.Q^T and O^T = V^T.P so each softmax row
// (query) is a single lane column: reductions are in-register + one
// shfl_xor(16); the O rescale is a per-lane scalar. Score C-tiles become the
// next WMMA's B operand with two shfl_xor(16) per register (no LDS, no
// barriers: one wave owns the whole tile).
// ---------------------------------------------------------------------------
__global__ __launch_bounds__(256) void k_attn(const bf16* __restrict__ Q,   // [B*S][HD]
                                              const bf16* __restrict__ Kp,  // [B*S][GD]
                                              const bf16* __restrict__ Vt,  // [GD][B*S]
                                              bf16* __restrict__ attn) {    // [B*S][HD]
  int lane = threadIdx.x & 31;
  int wave = threadIdx.x >> 5;
  int lh = lane >> 4, ln = lane & 15;

  int task = blockIdx.x * 8 + wave;          // B * (G*2) * (S/16) = 4096 tasks
  int qt = task & 127;                       // query tile
  int gp = (task >> 7) & 15;                 // g*2 + head-pair index
  int b  = task >> 11;
  int g  = gp >> 1;
  int h0 = g * 4 + (gp & 1) * 2;             // first of two heads in this task
  int q0 = qt * 16;
  int qg = q0 + ln;                          // this lane's query index
  const float scale = 0.125f;                // 1/sqrt(64)

  // Q fragments for both heads (B operand, lane = query column)
  v16bf bq[2][2];
#pragma unroll
  for (int hh = 0; hh < 2; ++hh) {
    const bf16* qrow = Q + (size_t)(b * kS + qg) * kHD + (h0 + hh) * kD;
    bq[hh][0] = load_fragB(qrow + 0 + 16 * lh);
    bq[hh][1] = load_fragB(qrow + 32 + 16 * lh);
  }

  v8f o[2][4];
#pragma unroll
  for (int hh = 0; hh < 2; ++hh)
#pragma unroll
    for (int t = 0; t < 4; ++t)
#pragma unroll
      for (int r = 0; r < 8; ++r) o[hh][t][r] = 0.0f;

  float mrun[2] = {-__builtin_inff(), -__builtin_inff()};
  float lrun[2] = {0.0f, 0.0f};

  int ktend = (q0 + 16 + 31) >> 5;           // 32-key tiles under causal mask
  for (int kt = 0; kt < ktend; ++kt) {
    int t0 = kt * 32;
    bool edge = (t0 + 31) > q0;

    // K A-fragments, loaded ONCE, used by both heads' score WMMAs
    const bf16* krow0 = Kp + (size_t)(b * kS + t0 + ln) * kGD + g * kD;
    const bf16* krow1 = krow0 + (size_t)16 * kGD;
    v16bf ka0 = load_fragA(krow0, 0, lh);
    v16bf ka1 = load_fragA(krow0, 32, lh);
    v16bf ka2 = load_fragA(krow1, 0, lh);
    v16bf ka3 = load_fragA(krow1, 32, lh);

    v16bf pb[2];
#pragma unroll
    for (int hh = 0; hh < 2; ++hh) {
      // scores^T: 2 C-tiles (keys t0..+15, t0+16..+31) x 16 queries
      v8f s0, s1;
#pragma unroll
      for (int r = 0; r < 8; ++r) { s0[r] = 0.0f; s1[r] = 0.0f; }
      s0 = wmma_bf16(ka0, bq[hh][0], s0);
      s0 = wmma_bf16(ka1, bq[hh][1], s0);
      s1 = wmma_bf16(ka2, bq[hh][0], s1);
      s1 = wmma_bf16(ka3, bq[hh][1], s1);

      // scale + causal mask (tile straddles the diagonal only near the end)
      float tmax = -__builtin_inff();
#pragma unroll
      for (int r = 0; r < 8; ++r) {
        float a = s0[r] * scale;
        float c = s1[r] * scale;
        if (edge) {
          int key0 = t0 + 8 * lh + r;
          if (key0 > qg)      a = -__builtin_inff();
          if (key0 + 16 > qg) c = -__builtin_inff();
        }
        s0[r] = a; s1[r] = c;
        tmax = fmaxf(tmax, fmaxf(a, c));
      }
      tmax = fmaxf(tmax, __shfl_xor(tmax, 16, 32));

      float mnew = fmaxf(mrun[hh], tmax);
      float fac = __expf(mrun[hh] - mnew);
      float psum = 0.0f;
#pragma unroll
      for (int r = 0; r < 8; ++r) {
        s0[r] = __expf(s0[r] - mnew);
        s1[r] = __expf(s1[r] - mnew);
        psum += s0[r] + s1[r];
      }
      psum += __shfl_xor(psum, 16, 32);
      lrun[hh] = lrun[hh] * fac + psum;
      mrun[hh] = mnew;
#pragma unroll
      for (int t = 0; t < 4; ++t)
#pragma unroll
        for (int r = 0; r < 8; ++r) o[hh][t][r] *= fac;

      // Build P as B operand (K dim = 32 keys). B needs key = t0 + e + 16*lh;
      // C-tiles hold key = t0 + r + 8*lh (+16 for s1) -> two half-swaps / reg.
#pragma unroll
      for (int r = 0; r < 8; ++r) {
        float x0 = __shfl_xor(s0[r], 16, 32);
        float x1 = __shfl_xor(s1[r], 16, 32);
        pb[hh][r]     = (bf16)(lh ? x1 : s0[r]);
        pb[hh][r + 8] = (bf16)(lh ? s1[r] : x0);
      }
    }

    // O^T += V^T x P : V A-fragments loaded ONCE, used by both heads
#pragma unroll
    for (int dt = 0; dt < 4; ++dt) {
      const bf16* vrow = Vt + (size_t)(g * kD + dt * 16 + ln) * kM + b * kS;
      v16bf va = load_fragA(vrow, t0, lh);
      o[0][dt] = wmma_bf16(va, pb[0], o[0][dt]);
      o[1][dt] = wmma_bf16(va, pb[1], o[1][dt]);
    }
  }

  // normalize + store: lane column is query qg; d = dt*16 + 8*lh + r contiguous
#pragma unroll
  for (int hh = 0; hh < 2; ++hh) {
    float rcp = 1.0f / lrun[hh];
    bf16* orow = attn + (size_t)(b * kS + qg) * kHD + (h0 + hh) * kD;
#pragma unroll
    for (int dt = 0; dt < 4; ++dt) {
      bf16x8 pk;
#pragma unroll
      for (int r = 0; r < 8; ++r) pk[r] = (bf16)(o[hh][dt][r] * rcp);
      *(bf16x8*)(orow + dt * 16 + 8 * lh) = pk;
    }
  }
}

// ---------------------------------------------------------------------------
extern "C" void kernel_launch(void* const* d_in, const int* in_sizes, int n_in,
                              void* d_out, int out_size, void* d_ws, size_t ws_size,
                              hipStream_t stream) {
  const float* x  = (const float*)d_in[0];
  const float* Wq = (const float*)d_in[1];
  const float* bq = (const float*)d_in[2];
  const float* Wk = (const float*)d_in[3];
  const float* bk = (const float*)d_in[4];
  const float* Wv = (const float*)d_in[5];
  const float* bv = (const float*)d_in[6];
  const float* Wo = (const float*)d_in[7];
  const float* bo = (const float*)d_in[8];
  float* out = (float*)d_out;

  char* ws = (char*)d_ws;
  size_t off = 0;
  auto alloc = [&](size_t bytes) -> char* {
    char* p = ws + off;
    off = (off + bytes + 255) & ~(size_t)255;
    return p;
  };
  bf16* xb   = (bf16*)alloc((size_t)kM * kDM * 2);   // x in bf16
  bf16* WqT  = (bf16*)alloc((size_t)kHD * kDM * 2);  // [N][K]
  bf16* WkT  = (bf16*)alloc((size_t)kGD * kDM * 2);
  bf16* WvT  = (bf16*)alloc((size_t)kGD * kDM * 2);
  bf16* WoT  = (bf16*)alloc((size_t)kDM * kHD * 2);
  bf16* Qb   = (bf16*)alloc((size_t)kM * kHD * 2);   // [B*S][HD]
  bf16* Kb   = (bf16*)alloc((size_t)kM * kGD * 2);   // [B*S][GD]
  bf16* Vt   = (bf16*)alloc((size_t)kGD * kM * 2);   // [GD][B*S]  (transposed)
  bf16* Ab   = (bf16*)alloc((size_t)kM * kHD * 2);   // attention output bf16

  // 1) convert x to bf16
  k_cvt<<<dim3((kM * kDM) / 1024), 256, 0, stream>>>(x, xb, kM * kDM);

  // 2) transpose weights to [N][K] bf16
  k_transpose<<<dim3(kHD / 32, kDM / 32), dim3(32, 8), 0, stream>>>(Wq, WqT, kDM, kHD);
  k_transpose<<<dim3(kGD / 32, kDM / 32), dim3(32, 8), 0, stream>>>(Wk, WkT, kDM, kGD);
  k_transpose<<<dim3(kGD / 32, kDM / 32), dim3(32, 8), 0, stream>>>(Wv, WvT, kDM, kGD);
  k_transpose<<<dim3(kDM / 32, kHD / 32), dim3(32, 8), 0, stream>>>(Wo, WoT, kHD, kDM);

  // 3) projections (V written transposed for the attention PV matmul)
  k_gemm<0><<<dim3(kHD / 64, kM / 128), 128, 0, stream>>>(xb, WqT, bq, Qb, kM, kHD, kDM);
  k_gemm<0><<<dim3(kGD / 64, kM / 128), 128, 0, stream>>>(xb, WkT, bk, Kb, kM, kGD, kDM);
  k_gemm<1><<<dim3(kGD / 64, kM / 128), 128, 0, stream>>>(xb, WvT, bv, Vt, kM, kGD, kDM);

  // 4) causal GQA flash attention: B*(G*2)*(S/16) wave tasks, 8 waves/block
  k_attn<<<dim3((kB * kG * 2 * (kS / 16)) / 8), 256, 0, stream>>>(Qb, Kb, Vt, Ab);

  // 5) output projection -> fp32 d_out
  k_gemm<2><<<dim3(kDM / 64, kM / 128), 128, 0, stream>>>(Ab, WoT, bo, out, kM, kDM, kHD);

  (void)in_sizes; (void)n_in; (void)out_size; (void)ws_size;
}
